// MambaSCAModel_81844896792905
// MI455X (gfx1250) — matrix operand order repaired and checked by reading, not compiled, not run on hardware
//
#include <hip/hip_runtime.h>
#include <hip/hip_bf16.h>

// ---------------------------------------------------------------------------
// Problem constants (from reference)
// ---------------------------------------------------------------------------
#define B_   256
#define T_   6000
#define DM   64      // model dim
#define DI   128     // inner dim
#define NN   8       // state size
#define KK   3       // stem conv kernel / stride
#define DC   4       // depthwise conv width
#define RR   4       // dt rank
#define NC_  256     // classes
#define LL   2000    // (T-K)/K + 1
#define TT   80      // time tile (multiple of 16, divides LL: 25 tiles)
#define NTILE (LL / TT)

// bf16 weight workspace layout (per direction p x block i, pi = p*2+i)
#define WIN_ELEMS  (256 * 64)   // in-proj  (2*Di rows x Dm cols)
#define WXP_ELEMS  (32 * 128)   // x-proj padded to 32 rows x Di cols
#define WOUT_ELEMS (64 * 128)   // out-proj (Dm rows x Di cols)
#define PI_STRIDE  (WIN_ELEMS + WXP_ELEMS + WOUT_ELEMS)  // 28672 bf16
#define HFEAT_OFF_BYTES (4 * PI_STRIDE * 2)              // after 4 weight sets

typedef __attribute__((ext_vector_type(16))) __bf16 v16bf;
typedef __attribute__((ext_vector_type(8)))  float  v8f;

__device__ __forceinline__ unsigned short f2bf(float f) {
  unsigned int u = __float_as_uint(f);
  u += 0x7fffu + ((u >> 16) & 1u);   // round-to-nearest-even
  return (unsigned short)(u >> 16);
}
__device__ __forceinline__ unsigned int pack_bf2(float lo, float hi) {
  return (unsigned int)f2bf(lo) | ((unsigned int)f2bf(hi) << 16);
}
__device__ __forceinline__ float sigm(float x) { return 1.f / (1.f + __expf(-x)); }

union Frag { unsigned int u[8]; v16bf v; };

// One 16x16 output tile of A(MxK,bf16,row-major,LDS) * W(NxK,bf16,row-major)^T.
// A fragment layout (ISA 7.12.2, 16-bit A 16x32): lane l -> row m0+(l&15),
//   K half (l>>4)*8; VGPR v -> K = half + (v>>2)*16 + (v&3)*2 (+0,+1 packed).
// B fragment layout (16-bit B 32x16): lane l -> col n0+(l&15), K half (l>>4)*8;
//   VGPR v -> K = half + (v>>2)*16 + (v&3)*2 (+0,+1 packed).
__device__ __forceinline__ v8f wmma_ktile(const unsigned short* A, int lda,
                                          const unsigned short* W, int ldw,
                                          int m0, int n0, int K, v8f c, int lane) {
  const int mr  = m0 + (lane & 15);
  const int nr  = n0 + (lane & 15);
  const int kbA = (lane >> 4) * 8;
  const int kbB = (lane >> 4) * 16;
  for (int k0 = 0; k0 < K; k0 += 32) {
    Frag a, b;
#pragma unroll
    for (int v = 0; v < 8; ++v) {
      const int ka = k0 + kbA + ((v >> 2) << 4) + ((v & 3) << 1);
      a.u[v] = *(const unsigned int*)(A + mr * lda + ka);
      const int kb = k0 + kbA + ((v >> 2) << 4) + ((v & 3) << 1);
      b.u[v] = *(const unsigned int*)(W + nr * ldw + kb);
    }
    c = __builtin_amdgcn_wmma_f32_16x16x32_bf16(false, a.v, false, b.v,
                                                (short)0, c, false, false);
  }
  return c;
}

// ---------------------------------------------------------------------------
// Kernel 1: repack fp32 weights -> bf16 workspace (x-proj padded 20->32 rows)
// ---------------------------------------------------------------------------
__global__ __launch_bounds__(256) void wconv_kernel(
    const float* f_in_w, const float* f_xp_w, const float* f_out_w,
    const float* b_in_w, const float* b_xp_w, const float* b_out_w,
    unsigned short* dst, int total) {
  int g = blockIdx.x * 256 + threadIdx.x;
  if (g >= total) return;
  int pi = g / PI_STRIDE;
  int r  = g - pi * PI_STRIDE;
  const float* in_w  = (pi < 2) ? f_in_w  : b_in_w;
  const float* xp_w  = (pi < 2) ? f_xp_w  : b_xp_w;
  const float* out_w = (pi < 2) ? f_out_w : b_out_w;
  const int blk = pi & 1;
  float v;
  if (r < WIN_ELEMS) {
    v = in_w[blk * WIN_ELEMS + r];
  } else if (r < WIN_ELEMS + WXP_ELEMS) {
    int rr = r - WIN_ELEMS, row = rr >> 7, col = rr & 127;
    v = (row < (RR + 2 * NN)) ? xp_w[blk * (RR + 2 * NN) * DI + row * DI + col] : 0.f;
  } else {
    int rr = r - WIN_ELEMS - WXP_ELEMS;
    v = out_w[blk * WOUT_ELEMS + rr];
  }
  dst[(size_t)pi * PI_STRIDE + r] = f2bf(v);
}

// ---------------------------------------------------------------------------
// Kernel 2: fused bidirectional Mamba, one workgroup per (batch, direction)
// ---------------------------------------------------------------------------
struct FP {
  const float* x;
  const float* conv_w;
  const float* conv_b;
  const float* ln_w[2];
  const float* ln_b[2];
  const float* cv_w[2];
  const float* cv_b[2];
  const float* dt_w[2];
  const float* dt_b[2];
  const float* Alog[2];
  const float* Dp[2];
  const unsigned short* wbf;
  float* hfeat;
};

__global__ __launch_bounds__(256) void mamba_fused(FP P) {
  const int b    = blockIdx.x;
  const int p    = blockIdx.y;   // 0 = forward, 1 = backward (reversed time)
  const int tid  = threadIdx.x;
  const int lane = tid & 31;
  const int wave = tid >> 5;

  extern __shared__ float smem[];
  float* xbuf = smem;                 // residual stream  TT x DM  (f32)
  float* xraw = xbuf + TT * DM;       // pre/post conv xi TT x DI  (f32, in place)
  float* zgb  = xraw + TT * DI;       // gate branch      TT x DI  (f32)
  float* dts  = zgb  + TT * DI;       // dt               TT x DI  (f32)
  float* xdb  = dts  + TT * DI;       // xdbl padded      TT x 32  (f32)
  unsigned short* u16  = (unsigned short*)(xdb + TT * 32);  // LN out   TT x DM (bf16)
  unsigned short* xi16 = (unsigned short*)(u16 + TT * DM);  // xi / y   TT x DI (bf16)

  // Per-channel persistent state (threads 0..127 each own channel d)
  float Areg[2][NN], Dreg[2], cw[2][DC], cb[2], tail[2][DC - 1], hst[2][NN];
  if (tid < DI) {
#pragma unroll
    for (int blk = 0; blk < 2; ++blk) {
#pragma unroll
      for (int n = 0; n < NN; ++n) {
        Areg[blk][n] = -__expf(P.Alog[p][blk * DI * NN + tid * NN + n]);
        hst[blk][n]  = 0.f;
      }
      Dreg[blk] = P.Dp[p][blk * DI + tid];
#pragma unroll
      for (int j = 0; j < DC; ++j) cw[blk][j] = P.cv_w[p][blk * DI * DC + tid * DC + j];
      cb[blk] = P.cv_b[p][blk * DI + tid];
      tail[blk][0] = tail[blk][1] = tail[blk][2] = 0.f;
    }
  }

  for (int tb = 0; tb < NTILE; ++tb) {
    // ---- stem conv (stride-3 VALID), directly from x, reversed for p==1 ----
    for (int i = tid; i < TT * DM; i += 256) {
      int t = i >> 6, c = i & (DM - 1);
      int gl = (p == 0) ? (tb * TT + t) : (LL - 1 - (tb * TT + t));
      const float* xp = P.x + (size_t)b * T_ + gl * KK;
      float acc = P.conv_b[c];
#pragma unroll
      for (int k = 0; k < KK; ++k) acc += P.conv_w[c * KK + k] * xp[k];
      xbuf[t * DM + c] = acc;
    }
    __syncthreads();

#pragma unroll
    for (int blk = 0; blk < 2; ++blk) {
      const unsigned short* wpi  = P.wbf + (size_t)(p * 2 + blk) * PI_STRIDE;
      const unsigned short* Win  = wpi;
      const unsigned short* Wxp  = wpi + WIN_ELEMS;
      const unsigned short* Wout = wpi + WIN_ELEMS + WXP_ELEMS;

      // ---- LayerNorm over DM, emit bf16 ----
      if (tid < TT) {
        const int t = tid;
        float s = 0.f, q = 0.f;
        for (int c = 0; c < DM; ++c) { float v = xbuf[t * DM + c]; s += v; q += v * v; }
        float m = s * (1.f / DM);
        float r = rsqrtf(q * (1.f / DM) - m * m + 1e-5f);
        for (int c = 0; c < DM; ++c) {
          float v = (xbuf[t * DM + c] - m) * r * P.ln_w[p][blk * DM + c]
                    + P.ln_b[p][blk * DM + c];
          u16[t * DM + c] = f2bf(v);
        }
      }
      __syncthreads();

      // ---- MM1: xz = u @ in_w^T  (TT x 256), split into xi / gate ----
      // The xi/gate split is WAVE-UNIFORM per 16x16 tile (DI % 16 == 0):
      // tiles with nt < 8 land wholly in xraw, nt >= 8 wholly in zgb.
      {
        const int ntN = 256 / 16, nT = (TT / 16) * ntN;  // 80 tiles
        for (int ti = wave; ti < nT; ti += 8) {
          int mt = ti / ntN, nt = ti % ntN;
          v8f c = {};
          c = wmma_ktile(u16, DM, Win, DM, mt * 16, nt * 16, DM, c, lane);
          // uniform destination selection (no per-lane divergence)
          float* dst = (nt < 8) ? (xraw + (nt * 16))
                                : (zgb + ((nt - 8) * 16));
          const int col = lane & 15;
          const int rb  = mt * 16 + 8 * (lane >> 4);
#pragma unroll
          for (int v = 0; v < 8; ++v) dst[(rb + v) * DI + col] = c[v];
        }
      }
      __syncthreads();

      // ---- depthwise causal conv (DC=4) + SiLU, carry tail across tiles ----
      if (tid < DI) {
        const int d = tid;
        float t0 = tail[blk][0], t1 = tail[blk][1], t2 = tail[blk][2];
        const float c0 = cw[blk][0], c1 = cw[blk][1], c2 = cw[blk][2], c3 = cw[blk][3];
        for (int t = 0; t < TT; ++t) {
          float cur = xraw[t * DI + d];
          float s = cb[blk] + c0 * t0 + c1 * t1 + c2 * t2 + c3 * cur;
          t0 = t1; t1 = t2; t2 = cur;
          float a = s * sigm(s);
          xraw[t * DI + d] = a;          // overwrite in place (safe: rolling regs)
          xi16[t * DI + d] = f2bf(a);
        }
        tail[blk][0] = t0; tail[blk][1] = t1; tail[blk][2] = t2;
      }
      __syncthreads();

      // ---- MM2: xdbl = xi @ xp_w^T  (TT x 32, rows 20..31 are zero-padded) ----
      {
        const int ntN = 2, nT = (TT / 16) * ntN;  // 10 tiles
        for (int ti = wave; ti < nT; ti += 8) {
          int mt = ti / ntN, nt = ti % ntN;
          v8f c = {};
          c = wmma_ktile(xi16, DI, Wxp, DI, mt * 16, nt * 16, DI, c, lane);
          int col = nt * 16 + (lane & 15);
          int rb  = mt * 16 + 8 * (lane >> 4);
#pragma unroll
          for (int v = 0; v < 8; ++v) xdb[(rb + v) * 32 + col] = c[v];
        }
      }
      __syncthreads();

      // ---- dt = softplus(dtr @ dt_w^T + dt_b) ----
      for (int i = tid; i < TT * DI; i += 256) {
        int t = i >> 7, d = i & (DI - 1);
        float v = P.dt_b[p][blk * DI + d];
#pragma unroll
        for (int r = 0; r < RR; ++r)
          v += xdb[t * 32 + r] * P.dt_w[p][blk * DI * RR + d * RR + r];
        dts[i] = (v > 20.f) ? v : log1pf(__expf(v));
      }
      __syncthreads();

      // ---- selective scan (state in regs) + SiLU gate, emit y as bf16 ----
      if (tid < DI) {
        const int d = tid;
        for (int t = 0; t < TT; ++t) {
          float dtv = dts[t * DI + d];
          float xiv = xraw[t * DI + d];
          float dx  = dtv * xiv;
          float y   = 0.f;
#pragma unroll
          for (int n = 0; n < NN; ++n) {
            float e = __expf(dtv * Areg[blk][n]);
            float h = hst[blk][n] * e + dx * xdb[t * 32 + RR + n];
            hst[blk][n] = h;
            y += h * xdb[t * 32 + RR + NN + n];
          }
          y += xiv * Dreg[blk];
          float z = zgb[t * DI + d];
          y *= z * sigm(z);
          xi16[t * DI + d] = f2bf(y);
        }
      }
      __syncthreads();

      // ---- MM3: residual += y @ out_w^T  (TT x 64), accumulate into xbuf ----
      {
        const int ntN = DM / 16, nT = (TT / 16) * ntN;  // 20 tiles
        for (int ti = wave; ti < nT; ti += 8) {
          int mt = ti / ntN, nt = ti % ntN;
          int col = nt * 16 + (lane & 15);
          int rb  = mt * 16 + 8 * (lane >> 4);
          v8f c;
#pragma unroll
          for (int v = 0; v < 8; ++v) c[v] = xbuf[(rb + v) * DM + col];
          c = wmma_ktile(xi16, DI, Wout, DI, mt * 16, nt * 16, DI, c, lane);
#pragma unroll
          for (int v = 0; v < 8; ++v) xbuf[(rb + v) * DM + col] = c[v];
        }
      }
      __syncthreads();
    }  // blk
  }    // tb

  // last streamed timestep (global index L-1 since LL % TT == 0)
  if (tid < DM)
    P.hfeat[(size_t)b * (2 * DM) + p * DM + tid] = xbuf[(TT - 1) * DM + tid];
}

// ---------------------------------------------------------------------------
// Kernel 3: classifier  out = h(256x128) @ cls_w^T(128x256) + b, WMMA bf16
// ---------------------------------------------------------------------------
__global__ __launch_bounds__(32) void cls_kernel(const float* hfeat,
                                                 const float* cls_w,
                                                 const float* cls_b,
                                                 float* out) {
  const int lane = threadIdx.x;
  const int n0 = blockIdx.x * 16, m0 = blockIdx.y * 16;
  const int mr  = m0 + (lane & 15);
  const int nr  = n0 + (lane & 15);
  const int kbA = (lane >> 4) * 8;
  const int kbB = (lane >> 4) * 16;
  v8f c = {};
  for (int k0 = 0; k0 < 2 * DM; k0 += 32) {
    Frag a, b;
#pragma unroll
    for (int v = 0; v < 8; ++v) {
      const int ka = k0 + kbA + ((v >> 2) << 4) + ((v & 3) << 1);
      a.u[v] = pack_bf2(hfeat[mr * 128 + ka], hfeat[mr * 128 + ka + 1]);
      const int kb = k0 + kbA + ((v >> 2) << 4) + ((v & 3) << 1);
      b.u[v] = pack_bf2(cls_w[nr * 128 + kb], cls_w[nr * 128 + kb + 1]);
    }
    c = __builtin_amdgcn_wmma_f32_16x16x32_bf16(false, a.v, false, b.v,
                                                (short)0, c, false, false);
  }
  const int col = n0 + (lane & 15);
  const float bias = cls_b[col];
#pragma unroll
  for (int v = 0; v < 8; ++v) {
    const int row = m0 + 8 * (lane >> 4) + v;
    out[row * NC_ + col] = c[v] + bias;
  }
}

// ---------------------------------------------------------------------------
// Host entry
// ---------------------------------------------------------------------------
extern "C" void kernel_launch(void* const* d_in, const int* in_sizes, int n_in,
                              void* d_out, int out_size, void* d_ws, size_t ws_size,
                              hipStream_t stream) {
  (void)in_sizes; (void)n_in; (void)out_size; (void)ws_size;
  const float* x      = (const float*)d_in[0];
  const float* conv_w = (const float*)d_in[1];
  const float* conv_b = (const float*)d_in[2];
  // per-direction param blocks: ln_w ln_b in_w cv_w cv_b xp_w dt_w dt_b Alog D out_w
  const float* f[11]; const float* bw[11];
  for (int i = 0; i < 11; ++i) {
    f[i]  = (const float*)d_in[3 + i];
    bw[i] = (const float*)d_in[14 + i];
  }
  const float* cls_w = (const float*)d_in[25];
  const float* cls_b = (const float*)d_in[26];

  unsigned short* wbf = (unsigned short*)d_ws;
  float* hfeat = (float*)((char*)d_ws + HFEAT_OFF_BYTES);

  // 1) repack weights to bf16
  {
    const int total = 4 * PI_STRIDE;
    wconv_kernel<<<(total + 255) / 256, 256, 0, stream>>>(
        f[2], f[5], f[10], bw[2], bw[5], bw[10], wbf, total);
  }

  // 2) fused streaming Mamba (512 workgroups, 184KB LDS each)
  FP P;
  P.x = x; P.conv_w = conv_w; P.conv_b = conv_b;
  P.ln_w[0] = f[0];  P.ln_b[0] = f[1];
  P.cv_w[0] = f[3];  P.cv_b[0] = f[4];
  P.dt_w[0] = f[6];  P.dt_b[0] = f[7];
  P.Alog[0] = f[8];  P.Dp[0]   = f[9];
  P.ln_w[1] = bw[0]; P.ln_b[1] = bw[1];
  P.cv_w[1] = bw[3]; P.cv_b[1] = bw[4];
  P.dt_w[1] = bw[6]; P.dt_b[1] = bw[7];
  P.Alog[1] = bw[8]; P.Dp[1]   = bw[9];
  P.wbf = wbf; P.hfeat = hfeat;

  const size_t smem_bytes =
      (size_t)(TT * DM + 3 * TT * DI + TT * 32) * sizeof(float)   // f32 buffers
      + (size_t)(TT * DM + TT * DI) * sizeof(unsigned short);     // bf16 buffers
  mamba_fused<<<dim3(B_, 2), 256, smem_bytes, stream>>>(P);

  // 3) classifier
  cls_kernel<<<dim3(NC_ / 16, B_ / 16), 32, 0, stream>>>(hfeat, cls_w, cls_b,
                                                         (float*)d_out);
}
